// ActionHead_53747220742389
// MI455X (gfx1250) — matrix operand order repaired and checked by reading, not compiled, not run on hardware
//
#include <hip/hip_runtime.h>
#include <hip/hip_bf16.h>

typedef __attribute__((ext_vector_type(16))) _Float16 v16h;
typedef __attribute__((ext_vector_type(8)))  float    v8f;
typedef __attribute__((ext_vector_type(4)))  int      v4i;

// Problem constants
static constexpr int NB   = 16;     // batch segments
static constexpr int NPTS = 16384;  // total points
static constexpr int HID  = 512;
static constexpr int TLEN = 8;
static constexpr int OUTD = 218;

#if __has_builtin(__builtin_amdgcn_global_load_async_to_lds_b128) && \
    __has_builtin(__builtin_amdgcn_s_wait_asynccnt)
#define HAVE_ASYNC_LDS 1
#else
#define HAVE_ASYNC_LDS 0
#endif

#if HAVE_ASYNC_LDS
// 16-byte global -> LDS async copy (tracked by ASYNCcnt).
__device__ __forceinline__ void async_cp16(const char* gsrc, char* ldst) {
    __attribute__((address_space(1))) char* g1 =
        (__attribute__((address_space(1))) char*)(char*)gsrc;   // addrspacecast
    __attribute__((address_space(3))) char* l3 =
        (__attribute__((address_space(3))) char*)ldst;          // addrspacecast
    __builtin_amdgcn_global_load_async_to_lds_b128(
        (__attribute__((address_space(1))) v4i*)g1,
        (__attribute__((address_space(3))) v4i*)l3, 0, 0);
}
#endif

// ---------------------------------------------------------------------------
// K0: segment prefix offsets
// ---------------------------------------------------------------------------
__global__ void k_offsets(const int* __restrict__ counts, int* __restrict__ off) {
    if (threadIdx.x == 0) {
        int s = 0;
        for (int b = 0; b < NB; ++b) { off[b] = s; s += counts[b]; }
        off[NB] = s;
    }
}

// ---------------------------------------------------------------------------
// K1: repack hm_w1[:512,:] (f32 row-major, K x N) into per-lane WMMA
//     B-fragment layout, f16.  Fragment (kt, ct): lane l holds 16 contiguous
//     halves = W[kt*32 + 16*(l>>4) + i][ct*16 + (l&15)], i = 0..15.
//     kt-major: each 32-KB K-slice is contiguous (async-copy friendly).
// ---------------------------------------------------------------------------
__global__ void k_repack_w1(const float* __restrict__ w1, _Float16* __restrict__ wf16) {
    int g = blockIdx.x * 256 + threadIdx.x;     // 0 .. 262143
    int i    = g & 15;
    int lane = (g >> 4) & 31;
    int ct   = (g >> 9) & 31;
    int kt   = g >> 14;
    int k = kt * 32 + 16 * (lane >> 4) + i;
    int n = ct * 16 + (lane & 15);
    wf16[g] = (_Float16)w1[k * HID + n];
}

// ---------------------------------------------------------------------------
// K2: per-t bias vectors for both MLPs:
//     tbias[t][j] = hm_b1[j]  + sum_e traj[t][e]*hm_w1 [512+e][j]
//     actV [t][j] = act_b1[j] + sum_e traj[t][e]*act_w1[512+e][j]
// ---------------------------------------------------------------------------
__global__ void k_tbias(const float* __restrict__ traj,
                        const float* __restrict__ hm_w1, const float* __restrict__ hm_b1,
                        const float* __restrict__ act_w1, const float* __restrict__ act_b1,
                        float* __restrict__ tbias, float* __restrict__ actV) {
    int o = blockIdx.x * 256 + threadIdx.x;     // 0 .. 8191
    int which = o >> 12;                        // 0 -> tbias, 1 -> actV
    int r = o & 4095;
    int t = r >> 9, j = r & 511;
    const float* w1 = which ? act_w1 : hm_w1;
    float s = which ? act_b1[j] : hm_b1[j];
    for (int e = 0; e < 64; ++e)
        s += traj[t * 64 + e] * w1[(512 + e) * HID + j];
    (which ? actV : tbias)[t * HID + j] = s;
}

// ---------------------------------------------------------------------------
// K3: main fused kernel.
//   Stage 1: Z(64x512) = point_embeds tile @ W via v_wmma_f32_16x16x32_f16,
//            B K-slices double-buffered global->LDS via async copies; the
//            per-fragment LDS reads use two live buffers (even/odd) so loads
//            overlap WMMA execution.
//   Stage 2: hm[n,t,0:4] = lrelu(Z[n]+tbias[t]) @ hm_w2 + hm_b2, from LDS.
// Block = 256 threads (8 waves); wave w: row strip (w&3), col half (w>>2).
// Dynamic LDS: zt 512*68*4 = 139264 B, then 2 x 32768 B B-slice buffers.
// ---------------------------------------------------------------------------
__global__ __launch_bounds__(256) void k_hm_gemm(
    const float* __restrict__ pe, const _Float16* __restrict__ wf16,
    const float* __restrict__ tbias, const float* __restrict__ w2,
    const float* __restrict__ b2, float* __restrict__ hm_out)
{
    extern __shared__ float zt[];               // [512 cols][stride 68] floats
    _Float16* lds_b = (_Float16*)(zt + 512 * 68);   // 2 x 16384 halves

    const int tid    = threadIdx.x;
    const int lane   = tid & 31;
    const int w      = tid >> 5;
    const int rstrip = w & 3;
    const int chalf  = w >> 2;
    const int hi     = lane >> 4;               // K-half selector
    const int nlane  = lane & 15;
    const int blk    = blockIdx.x;
    const int row    = blk * 64 + rstrip * 16 + nlane;

    v8f acc[16];
#pragma unroll
    for (int i = 0; i < 16; ++i)
#pragma unroll
        for (int v = 0; v < 8; ++v) acc[i][v] = 0.0f;

#if HAVE_ASYNC_LDS
    // Each thread streams 128 B (8 x b128) of the 32-KB K-slice.
    const char* wsrc = (const char*)wf16 + (size_t)tid * 128;
    char*       wdst = (char*)lds_b + (size_t)tid * 128;
    {   // prologue: slice 0 in flight
#pragma unroll
        for (int c = 0; c < 8; ++c) async_cp16(wsrc + c * 16, wdst + c * 16);
    }
#endif

    for (int kt = 0; kt < 16; ++kt) {
        const int kk = kt * 32;

#if HAVE_ASYNC_LDS
        if (kt + 1 < 16) {
            const char* s_ = wsrc + (size_t)(kt + 1) * 32768;
            char*       d_ = wdst + (size_t)((kt + 1) & 1) * 32768;
#pragma unroll
            for (int c = 0; c < 8; ++c) async_cp16(s_ + c * 16, d_ + c * 16);
            __builtin_amdgcn_s_wait_asynccnt(8);    // older slice retired
        } else {
            __builtin_amdgcn_s_wait_asynccnt(0);    // last slice fully landed
        }
        __syncthreads();                        // slice kt visible to all waves
        const _Float16* bslice = lds_b + (size_t)(kt & 1) * 16384;
#else
        const _Float16* bslice = wf16 + (size_t)kt * 16384;
#endif
        const _Float16* bbase = bslice + ((size_t)chalf * 16 * 32 + lane) * 16;

        // A fragment: f32 loads + inline cvt to f16 (16-bit A layout, 16x32)
        const float4* ap = (const float4*)(pe + (size_t)row * HID + kk + 16 * hi);
        float4 f0 = ap[0], f1 = ap[1];          // K = kb .. kb+7
        float4 f2 = ap[2], f3 = ap[3];          // K = kb+8 .. kb+15
        v16h a;
        a[0]=(_Float16)f0.x;  a[1]=(_Float16)f0.y;  a[2]=(_Float16)f0.z;  a[3]=(_Float16)f0.w;
        a[4]=(_Float16)f1.x;  a[5]=(_Float16)f1.y;  a[6]=(_Float16)f1.z;  a[7]=(_Float16)f1.w;
        a[8]=(_Float16)f2.x;  a[9]=(_Float16)f2.y;  a[10]=(_Float16)f2.z; a[11]=(_Float16)f2.w;
        a[12]=(_Float16)f3.x; a[13]=(_Float16)f3.y; a[14]=(_Float16)f3.z; a[15]=(_Float16)f3.w;

        // Two live B buffers (even/odd slots) so ds_loads overlap WMMAs.
        v16h beven = *(const v16h*)(bbase);     // fragment 0
#pragma unroll
        for (int ct = 0; ct < 16; ct += 2) {
            v16h bodd = *(const v16h*)(bbase + (size_t)(ct + 1) * 512);
            acc[ct] = __builtin_amdgcn_wmma_f32_16x16x32_f16(
                false, a, false, beven, (short)0, acc[ct], false, false);
            if (ct + 2 < 16)
                beven = *(const v16h*)(bbase + (size_t)(ct + 2) * 512);
            acc[ct + 1] = __builtin_amdgcn_wmma_f32_16x16x32_f16(
                false, a, false, bodd, (short)0, acc[ct + 1], false, false);
        }

#if HAVE_ASYNC_LDS
        __syncthreads();                        // done with slice kt before reuse
#endif
    }

    // Spill Z tile to LDS, column-major with stride 68 (bank-conflict free).
    const int rloc0 = rstrip * 16 + 8 * hi;     // C layout: VGPR v -> row v (+8 upper lanes)
#pragma unroll
    for (int ct = 0; ct < 16; ++ct) {
        const int col = chalf * 256 + ct * 16 + nlane;
        float* dst = zt + col * 68 + rloc0;
#pragma unroll
        for (int v = 0; v < 8; ++v) dst[v] = acc[ct][v];
    }
    __syncthreads();

    // Fused epilogue: 64 rows x 8 t  ->  hm (N, T, 4)
    for (int p = tid; p < 512; p += 256) {
        const int rloc = p & 63;
        const int t    = p >> 6;
        const float* tb = tbias + t * HID;
        float a0 = b2[0], a1 = b2[1], a2 = b2[2], a3 = b2[3];
        for (int j = 0; j < HID; ++j) {
            float hv = zt[j * 68 + rloc] + tb[j];
            float h  = hv >= 0.0f ? hv : 0.02f * hv;
            float4 wv = *(const float4*)(w2 + j * 4);
            a0 += h * wv.x; a1 += h * wv.y; a2 += h * wv.z; a3 += h * wv.w;
        }
        float* o = hm_out + ((size_t)(blk * 64 + rloc) * TLEN + t) * 4;
        o[0] = a0; o[1] = a1; o[2] = a2; o[3] = a3;
    }
}

// ---------------------------------------------------------------------------
// K4: per-segment stable softmax over points + weighted coord reduction -> xt
// ---------------------------------------------------------------------------
__global__ __launch_bounds__(256) void k_softmax_xt(
    const float* __restrict__ coords, const float* __restrict__ hm,
    const int* __restrict__ off, float* __restrict__ out)
{
    __shared__ float red[32 * 256];
    const int b = blockIdx.x, tid = threadIdx.x;
    const int s0 = off[b], s1 = off[b + 1];

    float lmax[8];
#pragma unroll
    for (int t = 0; t < 8; ++t) lmax[t] = -3.4e38f;
    for (int p = s0 + tid; p < s1; p += 256) {
        const float* hp = hm + (size_t)p * 32;
#pragma unroll
        for (int t = 0; t < 8; ++t) lmax[t] = fmaxf(lmax[t], hp[t * 4]);
    }
#pragma unroll
    for (int t = 0; t < 8; ++t) red[t * 256 + tid] = lmax[t];
    __syncthreads();
    for (int s = 128; s > 0; s >>= 1) {
        if (tid < s)
#pragma unroll
            for (int t = 0; t < 8; ++t)
                red[t * 256 + tid] = fmaxf(red[t * 256 + tid], red[t * 256 + tid + s]);
        __syncthreads();
    }
    float m[8];
#pragma unroll
    for (int t = 0; t < 8; ++t) m[t] = red[t * 256];
    __syncthreads();

    float lsum[8], lnum[8][3];
#pragma unroll
    for (int t = 0; t < 8; ++t) { lsum[t] = 0.f; lnum[t][0] = lnum[t][1] = lnum[t][2] = 0.f; }
    for (int p = s0 + tid; p < s1; p += 256) {
        const float* hp = hm + (size_t)p * 32;
        float c0 = coords[p * 3 + 0], c1 = coords[p * 3 + 1], c2 = coords[p * 3 + 2];
#pragma unroll
        for (int t = 0; t < 8; ++t) {
            float e = expf(hp[t * 4] - m[t]);   // TEMP == 1
            lsum[t]    += e;
            lnum[t][0] += e * (c0 + hp[t * 4 + 1]);
            lnum[t][1] += e * (c1 + hp[t * 4 + 2]);
            lnum[t][2] += e * (c2 + hp[t * 4 + 3]);
        }
    }
#pragma unroll
    for (int t = 0; t < 8; ++t) {
        red[t * 256 + tid] = lsum[t];
        red[(8 + t * 3 + 0) * 256 + tid] = lnum[t][0];
        red[(8 + t * 3 + 1) * 256 + tid] = lnum[t][1];
        red[(8 + t * 3 + 2) * 256 + tid] = lnum[t][2];
    }
    __syncthreads();
    for (int s = 128; s > 0; s >>= 1) {
        if (tid < s)
            for (int q = 0; q < 32; ++q)
                red[q * 256 + tid] += red[q * 256 + tid + s];
        __syncthreads();
    }
    if (tid < 24) {
        int t = tid / 3, c = tid % 3;
        out[(b * TLEN + t) * 3 + c] = red[(8 + t * 3 + c) * 256] / red[t * 256];
    }
}

// ---------------------------------------------------------------------------
// K5: per-segment max of point_embeds  (traj part of pc == traj_emb itself)
// ---------------------------------------------------------------------------
__global__ void k_maxpe(const float* __restrict__ pe, const int* __restrict__ off,
                        float* __restrict__ maxpe) {
    const int b = blockIdx.x;
    const int s0 = off[b], s1 = off[b + 1];
    for (int d = threadIdx.x; d < HID; d += 256) {
        float v = -3.4e38f;
        for (int p = s0; p < s1; ++p) v = fmaxf(v, pe[(size_t)p * HID + d]);
        maxpe[b * HID + d] = v;
    }
}

// ---------------------------------------------------------------------------
// K6: hidden2[b,t,:] = lrelu(maxpe[b] @ act_w1[:512] + actV[t])
// ---------------------------------------------------------------------------
__global__ void k_act_hidden(const float* __restrict__ maxpe, const float* __restrict__ act_w1,
                             const float* __restrict__ actV, float* __restrict__ hid2) {
    const int b = blockIdx.x;
    for (int j = threadIdx.x; j < HID; j += 256) {
        float u = 0.f;
        for (int k = 0; k < 512; ++k)
            u += maxpe[b * HID + k] * act_w1[(size_t)k * HID + j];
#pragma unroll
        for (int t = 0; t < 8; ++t) {
            float hv = u + actV[t * HID + j];
            hid2[(size_t)(b * TLEN + t) * HID + j] = hv >= 0.f ? hv : 0.02f * hv;
        }
    }
}

// ---------------------------------------------------------------------------
// K7: ae = hidden2 @ act_w2 + act_b2, scattered into (xr, xo, xstop)
// ---------------------------------------------------------------------------
__global__ __launch_bounds__(256) void k_act_out(
    const float* __restrict__ hid2, const float* __restrict__ act_w2,
    const float* __restrict__ act_b2, float* __restrict__ out) {
    __shared__ float hrow[HID];
    const int row = blockIdx.x;                 // b*8 + t, 0..127
    const int tid = threadIdx.x;
    hrow[tid]       = hid2[(size_t)row * HID + tid];
    hrow[tid + 256] = hid2[(size_t)row * HID + tid + 256];
    __syncthreads();
    if (tid < OUTD) {
        float acc = act_b2[tid];
        for (int j = 0; j < HID; ++j)
            acc += hrow[j] * act_w2[(size_t)j * OUTD + tid];
        // out layout: xt[384] | xr[128*216] | xo[128] | xstop[128]
        if (tid < 216)       out[384 + row * 216 + tid] = acc;      // xr flat
        else if (tid == 216) out[384 + 27648 + row]       = acc;    // xo
        else                 out[384 + 27648 + 128 + row] = acc;    // xstop
    }
}

// ---------------------------------------------------------------------------
extern "C" void kernel_launch(void* const* d_in, const int* in_sizes, int n_in,
                              void* d_out, int out_size, void* d_ws, size_t ws_size,
                              hipStream_t stream) {
    const float* pe      = (const float*)d_in[0];
    const float* coords  = (const float*)d_in[1];
    const float* traj    = (const float*)d_in[2];
    const float* hm_w1   = (const float*)d_in[3];
    const float* hm_b1   = (const float*)d_in[4];
    const float* hm_w2   = (const float*)d_in[5];
    const float* hm_b2   = (const float*)d_in[6];
    const float* act_w1  = (const float*)d_in[7];
    const float* act_b1  = (const float*)d_in[8];
    const float* act_w2  = (const float*)d_in[9];
    const float* act_b2  = (const float*)d_in[10];
    const int*   counts  = (const int*)d_in[11];
    float*       out     = (float*)d_out;

    char* ws = (char*)d_ws;                     // 128B-aligned regions
    int*       off   = (int*)(ws + 0);          //   128 B
    _Float16*  wf16  = (_Float16*)(ws + 128);   //   512 KB repacked weights
    float*     tbias = (float*)(ws + 524416);   //   16 KB
    float*     actV  = (float*)(ws + 540800);   //   16 KB
    float*     hm    = (float*)(ws + 557184);   //  2 MB  (N, T, 4)
    float*     maxpe = (float*)(ws + 2654336);  //   32 KB
    float*     hid2  = (float*)(ws + 2687104);  //  256 KB

    const size_t gemm_lds = 512 * 68 * 4 + 2 * 32768;   // zt + double-buffered B

    k_offsets   <<<1,    32, 0,      stream>>>(counts, off);
    k_repack_w1 <<<1024, 256, 0,     stream>>>(hm_w1, wf16);
    k_tbias     <<<32,   256, 0,     stream>>>(traj, hm_w1, hm_b1, act_w1, act_b1, tbias, actV);
    k_hm_gemm   <<<NPTS / 64, 256, gemm_lds, stream>>>(pe, wf16, tbias, hm_w2, hm_b2, hm);
    k_softmax_xt<<<NB,   256, 0,     stream>>>(coords, hm, off, out);
    k_maxpe     <<<NB,   256, 0,     stream>>>(pe, off, maxpe);
    k_act_hidden<<<NB,   256, 0,     stream>>>(maxpe, act_w1, actV, hid2);
    k_act_out   <<<NB * TLEN, 256, 0, stream>>>(hid2, act_w2, act_b2, out);
}
